// PQMatcher_53120155517066
// MI455X (gfx1250) — matrix operand order repaired and checked by reading, not compiled, not run on hardware
//
#include <hip/hip_runtime.h>

// ---------------- problem constants ----------------
constexpr int kLP = 256, kLQ = 256, kB = 32, kIN = 512, kH = 1024, kH3 = 3072;

// ---------------- types ----------------
typedef __attribute__((ext_vector_type(16))) __bf16 v16bf;
typedef __attribute__((ext_vector_type(8)))  float  v8f;
typedef __attribute__((ext_vector_type(4)))  float  v4f;

union BFVec { v16bf v; v4f f[2]; };

__device__ __forceinline__ unsigned short f2bf(float x) {
  unsigned u = __float_as_uint(x);
  return (unsigned short)((u + 0x7FFFu + ((u >> 16) & 1u)) >> 16);
}
__device__ __forceinline__ float sigm(float x) { return 1.0f / (1.0f + __expf(-x)); }

// ---------------- WMMA core: one wave computes a 16x64 strip ----------------
// A: [M x K] row-major bf16 (lane indexes M), B: [N x K] row-major bf16
// (weight rows; lane indexes N). One A fragment feeds 4 WMMAs (4 N-tiles).
// All 4 B fragments are loaded into distinct registers BEFORE the WMMAs so the
// backend can issue one load clause and overlap L2 latency with the matrix pipe.
__device__ __forceinline__ void wmma_strip4(const unsigned short* __restrict__ Arow,
                                            const unsigned short* __restrict__ Bbase,
                                            int ldb, int K, int kh, v8f acc[4]) {
  for (int kc = 0; kc < K; kc += 32) {
    BFVec a, b0, b1, b2, b3;
    // A 16-bit layout: VGPR0-7 <- K = kc + 16*kh .. +15 (contiguous per lane)
    a.f[0] = *(const v4f*)(Arow + kc + 16 * kh);
    a.f[1] = *(const v4f*)(Arow + kc + 16 * kh + 8);
    // B 16-bit layout: VGPR0-7 <- K = kc + 16*kh .. +15 (contiguous per lane)
    const unsigned short* Br0 = Bbase + kc + 16 * kh;
    b0.f[0] = *(const v4f*)(Br0);
    b0.f[1] = *(const v4f*)(Br0 + 8);
    b1.f[0] = *(const v4f*)(Br0 + (size_t)16 * ldb);
    b1.f[1] = *(const v4f*)(Br0 + (size_t)16 * ldb + 8);
    b2.f[0] = *(const v4f*)(Br0 + (size_t)32 * ldb);
    b2.f[1] = *(const v4f*)(Br0 + (size_t)32 * ldb + 8);
    b3.f[0] = *(const v4f*)(Br0 + (size_t)48 * ldb);
    b3.f[1] = *(const v4f*)(Br0 + (size_t)48 * ldb + 8);
    acc[0] = __builtin_amdgcn_wmma_f32_16x16x32_bf16(false, a.v, false, b0.v,
                                                     (short)0, acc[0], false, false);
    acc[1] = __builtin_amdgcn_wmma_f32_16x16x32_bf16(false, a.v, false, b1.v,
                                                     (short)0, acc[1], false, false);
    acc[2] = __builtin_amdgcn_wmma_f32_16x16x32_bf16(false, a.v, false, b2.v,
                                                     (short)0, acc[2], false, false);
    acc[3] = __builtin_amdgcn_wmma_f32_16x16x32_bf16(false, a.v, false, b3.v,
                                                     (short)0, acc[3], false, false);
  }
}

#define GEMM_PROLOG(Mtiles)                                            \
  int t    = blockIdx.x * 4 + (threadIdx.x >> 5);                      \
  int lane = threadIdx.x & 31;                                         \
  int idx  = lane & 15, kh = lane >> 4;                                \
  int mt = t % (Mtiles), nt4 = t / (Mtiles);                           \
  v8f acc[4] = {{0.f,0.f,0.f,0.f,0.f,0.f,0.f,0.f},                     \
                {0.f,0.f,0.f,0.f,0.f,0.f,0.f,0.f},                     \
                {0.f,0.f,0.f,0.f,0.f,0.f,0.f,0.f},                     \
                {0.f,0.f,0.f,0.f,0.f,0.f,0.f,0.f}};

// plain f32 store (WUp/WUq/vWv)
__global__ __launch_bounds__(128)
void gemm_f32(const unsigned short* __restrict__ A, int lda,
              const unsigned short* __restrict__ B, int ldb,
              float* __restrict__ C, int ldc, int Mtiles, int K) {
  GEMM_PROLOG(Mtiles)
  wmma_strip4(A + (mt * 16 + idx) * lda, B + (nt4 * 64 + idx) * ldb, ldb, K, kh, acc);
  #pragma unroll
  for (int j = 0; j < 4; ++j)
    #pragma unroll
    for (int r = 0; r < 8; ++r)
      C[(mt * 16 + r + 8 * kh) * ldc + nt4 * 64 + j * 16 + idx] = acc[j][r];
}

// gate GEMM epilogue: out_bf[b,n] = bf16( sigmoid(acc[b,n]) * ch[b, n&511] )
__global__ __launch_bounds__(128)
void gemm_sig_mul_bf16(const unsigned short* __restrict__ A, int lda,
                       const unsigned short* __restrict__ B, int ldb,
                       const float* __restrict__ ch,
                       unsigned short* __restrict__ outbf, int Mtiles, int K) {
  GEMM_PROLOG(Mtiles)
  wmma_strip4(A + (mt * 16 + idx) * lda, B + (nt4 * 64 + idx) * ldb, ldb, K, kh, acc);
  #pragma unroll
  for (int j = 0; j < 4; ++j) {
    int n = nt4 * 64 + j * 16 + idx;
    #pragma unroll
    for (int r = 0; r < 8; ++r) {
      int b = mt * 16 + r + 8 * kh;
      float v = sigm(acc[j][r]) * ch[b * kIN + (n & (kIN - 1))];
      outbf[b * kH + n] = f2bf(v);
    }
  }
}

// fused GRU projections: blockIdx.y==0 -> gi = cbf@Wih.T + bih
//                        blockIdx.y==1 -> gh = vbf@Whh.T + bhh
__global__ __launch_bounds__(128)
void gru_gemm_k(const unsigned short* __restrict__ A0, const unsigned short* __restrict__ A1,
                const unsigned short* __restrict__ B0, const unsigned short* __restrict__ B1,
                const float* __restrict__ bias0, const float* __restrict__ bias1,
                float* __restrict__ C0, float* __restrict__ C1, int Mtiles, int K) {
  const unsigned short* A = blockIdx.y ? A1 : A0;
  const unsigned short* B = blockIdx.y ? B1 : B0;
  const float* bias       = blockIdx.y ? bias1 : bias0;
  float* C                = blockIdx.y ? C1 : C0;
  GEMM_PROLOG(Mtiles)
  wmma_strip4(A + (mt * 16 + idx) * kH, B + (nt4 * 64 + idx) * kH, kH, K, kh, acc);
  #pragma unroll
  for (int j = 0; j < 4; ++j) {
    float bv = bias[nt4 * 64 + j * 16 + idx];
    #pragma unroll
    for (int r = 0; r < 8; ++r)
      C[(mt * 16 + r + 8 * kh) * kH3 + nt4 * 64 + j * 16 + idx] = acc[j][r] + bv;
  }
}

// ---------------- elementwise / conversion kernels ----------------
__global__ void cvt_bf_k(const float* __restrict__ x, unsigned short* __restrict__ o, int n) {
  int i = blockIdx.x * 256 + threadIdx.x;
  if (i < n) o[i] = f2bf(x[i]);
}
// W [rows x 1024] -> out [rows x 512], out[r,k] = W[r,k] + W[r,k+512]
__global__ void fold_half_k(const float* __restrict__ W, unsigned short* __restrict__ o, int n) {
  int i = blockIdx.x * 256 + threadIdx.x;
  if (i >= n) return;
  int r = i >> 9, k = i & 511;
  o[i] = f2bf(W[r * 1024 + k] + W[r * 1024 + 512 + k]);
}
// Wg [2048x2048]: keep rows 1024..2047, fold K 2048->1024 for r=[u,u,ch,ch]
__global__ void fold_wg_k(const float* __restrict__ Wg, unsigned short* __restrict__ o) {
  int i = blockIdx.x * 256 + threadIdx.x;        // 1024*1024
  int m = i >> 10, k = i & 1023;
  const float* row = Wg + (size_t)(1024 + m) * 2048;
  float v = (k < 512) ? (row[k] + row[k + 512])
                      : (row[1024 + (k - 512)] + row[1536 + (k - 512)]);
  o[i] = f2bf(v);
}
__global__ void init_v_k(const float* __restrict__ v0, float* __restrict__ v,
                         unsigned short* __restrict__ vbf) {
  int i = blockIdx.x * 256 + threadIdx.x;        // 32*1024
  float x = v0[i];
  v[i] = x; vbf[i] = f2bf(x);
}

// ---------------- attention score: one wave per (q,b) ----------------
__global__ __launch_bounds__(256)
void attn_score_k(const float* __restrict__ WUq, const float* __restrict__ WUp_p,
                  const float* __restrict__ vWv, const float* __restrict__ Vvec,
                  float* __restrict__ s) {
  int g = blockIdx.x * 8 + (threadIdx.x >> 5);   // 8192 waves
  int lane = threadIdx.x & 31;
  int q = g >> 5, b = g & 31;
  const float* wuq = WUq + (size_t)(q * kB + b) * kH;
  const float* wup = WUp_p + (size_t)b * kH;
  const float* vw  = vWv + (size_t)b * kH;
  const float* vv  = Vvec + (size_t)b * kH;
  float sum = 0.f;
  #pragma unroll 4
  for (int h = lane; h < kH; h += 32)
    sum += tanhf(wuq[h] + wup[h] + vw[h]) * vv[h];
  #pragma unroll
  for (int off = 16; off; off >>= 1) sum += __shfl_xor(sum, off, 32);
  if (lane == 0) s[b * kLQ + q] = sum;
}

// ---------------- softmax + context + gate-GEMM input assembly ----------------
// block b (32 blocks, 256 threads = one q per thread)
__global__ __launch_bounds__(256)
void softmax_ctx_k(const float* __restrict__ s, const float* __restrict__ uq,
                   const float* __restrict__ up_p,
                   float* __restrict__ ch, unsigned short* __restrict__ xg) {
  __shared__ float red[256];
  __shared__ float av[256];
  int b = blockIdx.x, q = threadIdx.x;
  float val = s[b * kLQ + q];
  red[q] = val; __syncthreads();
  for (int off = 128; off; off >>= 1) { if (q < off) red[q] = fmaxf(red[q], red[q + off]); __syncthreads(); }
  float m = red[0]; __syncthreads();
  float e = __expf(val - m);
  red[q] = e; __syncthreads();
  for (int off = 128; off; off >>= 1) { if (q < off) red[q] += red[q + off]; __syncthreads(); }
  av[q] = e / red[0];
  __syncthreads();
  // context half: ch[b,d] = sum_q a[q] * uq[q,b,d]; also build xg = bf16([up_i, ch])
  for (int d = q; d < kIN; d += 256) {
    float acc = 0.f;
    const float* uqb = uq + (size_t)b * kIN + d;
    #pragma unroll 4
    for (int qq = 0; qq < kLQ; ++qq)
      acc += av[qq] * uqb[(size_t)qq * kB * kIN];
    ch[b * kIN + d] = acc;
    xg[b * kH + d]       = f2bf(up_p[b * kIN + d]);
    xg[b * kH + kIN + d] = f2bf(acc);
  }
}

// ---------------- GRU gate combine ----------------
__global__ __launch_bounds__(256)
void gru_gate_k(const float* __restrict__ gi, const float* __restrict__ gh,
                float* __restrict__ v, unsigned short* __restrict__ vbf,
                float* __restrict__ out_p) {
  int i = blockIdx.x * 256 + threadIdx.x;        // 32*1024
  int b = i >> 10, h = i & 1023;
  const float* gib = gi + (size_t)b * kH3;
  const float* ghb = gh + (size_t)b * kH3;
  float r = sigm(gib[h] + ghb[h]);
  float z = sigm(gib[kH + h] + ghb[kH + h]);
  float n = tanhf(gib[2 * kH + h] + r * ghb[2 * kH + h]);
  float hp = v[i];
  float vn = (1.f - z) * n + z * hp;
  v[i] = vn; vbf[i] = f2bf(vn); out_p[i] = vn;
}

// ---------------- host ----------------
extern "C" void kernel_launch(void* const* d_in, const int* in_sizes, int n_in,
                              void* d_out, int out_size, void* d_ws, size_t ws_size,
                              hipStream_t stream) {
  (void)in_sizes; (void)n_in; (void)out_size; (void)ws_size;
  const float* up  = (const float*)d_in[0];
  const float* uq  = (const float*)d_in[1];
  const float* v0  = (const float*)d_in[2];
  const float* Vv  = (const float*)d_in[3];
  const float* Wp  = (const float*)d_in[4];
  const float* Wq  = (const float*)d_in[5];
  const float* Wv  = (const float*)d_in[6];
  const float* Wg  = (const float*)d_in[7];
  const float* Wih = (const float*)d_in[8];
  const float* Whh = (const float*)d_in[9];
  const float* bih = (const float*)d_in[10];
  const float* bhh = (const float*)d_in[11];
  float* out = (float*)d_out;

  char* w = (char*)d_ws;
  auto alloc = [&](size_t bytes) -> char* {
    char* p = w; w += (bytes + 255) & ~(size_t)255; return p;
  };
  const int MB = kLP * kB;                                   // 8192 rows
  unsigned short* up_bf  = (unsigned short*)alloc((size_t)MB * kIN * 2);
  unsigned short* uq_bf  = (unsigned short*)alloc((size_t)MB * kIN * 2);
  unsigned short* Wpf    = (unsigned short*)alloc((size_t)kH * kIN * 2);
  unsigned short* Wqf    = (unsigned short*)alloc((size_t)kH * kIN * 2);
  unsigned short* Wv_bf  = (unsigned short*)alloc((size_t)kH * kH * 2);
  unsigned short* Wgf    = (unsigned short*)alloc((size_t)kH * kH * 2);
  unsigned short* Wih_bf = (unsigned short*)alloc((size_t)kH3 * kH * 2);
  unsigned short* Whh_bf = (unsigned short*)alloc((size_t)kH3 * kH * 2);
  float*          WUp    = (float*)alloc((size_t)MB * kH * 4);
  float*          WUq    = (float*)alloc((size_t)MB * kH * 4);
  float*          vWv    = (float*)alloc((size_t)kB * kH * 4);
  float*          sbuf   = (float*)alloc((size_t)kB * kLQ * 4);
  float*          ch     = (float*)alloc((size_t)kB * kIN * 4);
  unsigned short* xg     = (unsigned short*)alloc((size_t)kB * kH * 2);
  unsigned short* cbf    = (unsigned short*)alloc((size_t)kB * kH * 2);
  float*          gi     = (float*)alloc((size_t)kB * kH3 * 4);
  float*          gh     = (float*)alloc((size_t)kB * kH3 * 4);
  float*          vcur   = (float*)alloc((size_t)kB * kH * 4);
  unsigned short* vbf    = (unsigned short*)alloc((size_t)kB * kH * 2);

  // ---- precompute: conversions / weight folds ----
  cvt_bf_k<<<(MB * kIN) / 256, 256, 0, stream>>>(up, up_bf, MB * kIN);
  cvt_bf_k<<<(MB * kIN) / 256, 256, 0, stream>>>(uq, uq_bf, MB * kIN);
  fold_half_k<<<(kH * kIN) / 256, 256, 0, stream>>>(Wp, Wpf, kH * kIN);
  fold_half_k<<<(kH * kIN) / 256, 256, 0, stream>>>(Wq, Wqf, kH * kIN);
  cvt_bf_k<<<(kH * kH) / 256, 256, 0, stream>>>(Wv, Wv_bf, kH * kH);
  fold_wg_k<<<(kH * kH) / 256, 256, 0, stream>>>(Wg, Wgf);
  cvt_bf_k<<<(kH3 * kH) / 256, 256, 0, stream>>>(Wih, Wih_bf, kH3 * kH);
  cvt_bf_k<<<(kH3 * kH) / 256, 256, 0, stream>>>(Whh, Whh_bf, kH3 * kH);
  init_v_k<<<(kB * kH) / 256, 256, 0, stream>>>(v0, vcur, vbf);

  // ---- precompute: WUp / WUq  (M=8192, N=1024, K=512), 16x64 strips ----
  {
    int Mt = MB / 16, tiles = Mt * (kH / 64);               // 512*16
    gemm_f32<<<tiles / 4, 128, 0, stream>>>(up_bf, kIN, Wpf, kIN, WUp, kH, Mt, kIN);
    gemm_f32<<<tiles / 4, 128, 0, stream>>>(uq_bf, kIN, Wqf, kIN, WUq, kH, Mt, kIN);
  }

  // ---- sequential scan ----
  for (int p = 0; p < kLP; ++p) {
    const float* WUp_p = WUp + (size_t)p * kB * kH;
    const float* up_p  = up + (size_t)p * kB * kIN;
    float* out_p       = out + (size_t)p * kB * kH;

    // vWv = v @ Wv.T  (M=32, N=1024, K=1024): 2*16 strips -> 8 blocks
    gemm_f32<<<(2 * (kH / 64)) / 4, 128, 0, stream>>>(vbf, kH, Wv_bf, kH, vWv, kH, 2, kH);
    // attention scores (8192 waves)
    attn_score_k<<<1024, 256, 0, stream>>>(WUq, WUp_p, vWv, Vv, sbuf);
    // softmax + context + assemble gate-GEMM input
    softmax_ctx_k<<<kB, 256, 0, stream>>>(sbuf, uq, up_p, ch, xg);
    // gate: c_ = sigmoid(xg @ Wgf.T) * [ch,ch]   (M=32, N=1024, K=1024)
    gemm_sig_mul_bf16<<<(2 * (kH / 64)) / 4, 128, 0, stream>>>(xg, kH, Wgf, kH, ch, cbf, 2, kH);
    // GRU projections fused in one launch: gi and gh  (M=32, N=3072, K=1024)
    dim3 ggrid((2 * (kH3 / 64)) / 4, 2);
    gru_gemm_k<<<ggrid, 128, 0, stream>>>(cbf, vbf, Wih_bf, Whh_bf, bih, bhh, gi, gh, 2, kH);
    // gate combine -> v_new (also writes vs[p])
    gru_gate_k<<<(kB * kH) / 256, 256, 0, stream>>>(gi, gh, vcur, vbf, out_p);
  }
}
